// GAGNN_R2_81544249082504
// MI455X (gfx1250) — matrix operand order repaired and checked by reading, not compiled, not run on hardware
//
#include <hip/hip_runtime.h>
#include <math.h>

// ---------------- model constants ----------------
#define DD 128          // feature dim
#define CHN 8           // multivector channels
#define SD (CHN*DD)     // 1024 floats per node state row
#define H5 (5*DD)       // 640
#define NRBF 20
#define NTYPES 5
#define NROUNDS 4
#define RCUTF 5.0f
#define ETILE 64        // edges per block (4 M-tiles of 16)

typedef __attribute__((ext_vector_type(16))) __bf16 v16bf;
typedef __attribute__((ext_vector_type(8)))  __bf16 v8bf;
typedef __attribute__((ext_vector_type(8)))  float  v8f;

static __device__ __forceinline__ float siluf(float x) { return x / (1.f + __expf(-x)); }

static __device__ __forceinline__ v8f wmma_bf16(v16bf a, v16bf b, v8f c) {
  return __builtin_amdgcn_wmma_f32_16x16x32_bf16(false, a, false, b, (short)0, c, false, false);
}

// Build a 16x32 A fragment from 4 float4 chunks (k = kbase..+7, kbase+16..+23).
static __device__ __forceinline__ v16bf cvt16(float4 f0, float4 f1, float4 f2, float4 f3) {
  v16bf a;
  a[0] = (__bf16)f0.x;  a[1] = (__bf16)f0.y;  a[2] = (__bf16)f0.z;  a[3] = (__bf16)f0.w;
  a[4] = (__bf16)f1.x;  a[5] = (__bf16)f1.y;  a[6] = (__bf16)f1.z;  a[7] = (__bf16)f1.w;
  a[8] = (__bf16)f2.x;  a[9] = (__bf16)f2.y;  a[10] = (__bf16)f2.z; a[11] = (__bf16)f2.w;
  a[12] = (__bf16)f3.x; a[13] = (__bf16)f3.y; a[14] = (__bf16)f3.z; a[15] = (__bf16)f3.w;
  return a;
}

// A fragment from LDS: two 16-byte chunks (k = kbase..+7 and kbase+16..+23). p 16B-aligned.
static __device__ __forceinline__ v16bf ldsfrag(const __bf16* p) {
  v8bf lo = *reinterpret_cast<const v8bf*>(p);
  v8bf hi = *reinterpret_cast<const v8bf*>(p + 16);
  return __builtin_shufflevector(lo, hi, 0, 1, 2, 3, 4, 5, 6, 7, 8, 9, 10, 11, 12, 13, 14, 15);
}

// ---------------- weight repack: fp32 KxN row-major -> bf16 WMMA B fragments ----------------
// Fragment (kt,nt) stored at dst[(nt*Kt + kt)*512 + lane*16 + e]
// B layout: lanes 0-15: K = kt*32 + (e<8 ? e : e+8), N = nt*16 + lane; lanes 16-31: K = kt*32 + 8 + (e<8 ? e : e+8).
__global__ void k_repack(const float* __restrict__ src, __bf16* __restrict__ dst, int K, int Nc) {
  int Kt = (K + 31) >> 5;
  int Nt = (Nc + 15) >> 4;
  int t = blockIdx.x * 256 + threadIdx.x;
  int frag = t >> 9;
  if (frag >= Kt * Nt) return;
  int r = t & 511;
  int nt = frag / Kt, kt = frag % Kt;
  int lane = r >> 4, e = r & 15;
  int k = kt * 32 + ((lane & 16) ? 8 : 0) + (e < 8 ? e : e + 8);
  int n = nt * 16 + (lane & 15);
  float v = (k < K && n < Nc) ? src[(size_t)k * Nc + n] : 0.f;
  dst[(size_t)frag * 512 + r] = (__bf16)v;
}

// ---------------- generic WMMA GEMM: C[MxN] = act(A[MxK] @ W + bias) ----------------
// Requires K % 32 == 0 and Nc % 64 == 0 (true for all call sites).
// One wave computes a 16x64 tile: one A fragment feeds 4 WMMAs.
__global__ __launch_bounds__(256) void k_gemm(
    const float* __restrict__ A, int lda,
    const __bf16* __restrict__ Wp,
    const float* __restrict__ bias,
    float* __restrict__ C, int ldc,
    int M, int K, int Nc, int act) {
  int Kt = K >> 5;
  int Nt4 = Nc >> 6;
  int Mt = (M + 15) >> 4;
  int lane = threadIdx.x & 31;
  int tile = blockIdx.x * 8 + (threadIdx.x >> 5);
  if (tile >= Mt * Nt4) return;
  int mt = tile / Nt4, n4 = tile % Nt4;

  int mrowA = mt * 16 + (lane & 15);
  int mclamp = (mrowA < M) ? mrowA : (M - 1);  // clamp: loads unconditional, stores masked
  const float* Arow = A + (size_t)mclamp * lda + ((lane & 16) ? 8 : 0);
  const __bf16* Wb = Wp + (size_t)(n4 * 4) * Kt * 512 + lane * 16;

  v8f acc[4];
#pragma unroll
  for (int j = 0; j < 4; ++j) acc[j] = (v8f){};

  for (int kt = 0; kt < Kt; ++kt) {
    const float* ap = Arow + kt * 32;
    float4 f0 = *reinterpret_cast<const float4*>(ap);
    float4 f1 = *reinterpret_cast<const float4*>(ap + 4);
    float4 f2 = *reinterpret_cast<const float4*>(ap + 16);
    float4 f3 = *reinterpret_cast<const float4*>(ap + 20);
    v16bf a = cvt16(f0, f1, f2, f3);
    const __bf16* wk = Wb + (size_t)kt * 512;
#pragma unroll
    for (int j = 0; j < 4; ++j) {
      v16bf b = *reinterpret_cast<const v16bf*>(wk + (size_t)j * Kt * 512);
      acc[j] = wmma_bf16(a, b, acc[j]);
    }
  }

#pragma unroll
  for (int j = 0; j < 4; ++j) {
    int ncol = n4 * 64 + j * 16 + (lane & 15);
    float bv = bias ? bias[ncol] : 0.f;
#pragma unroll
    for (int i = 0; i < 8; ++i) {
      int mrow = mt * 16 + i + ((lane & 16) ? 8 : 0);
      if (mrow >= M) continue;
      float v = acc[j][i] + bv;
      if (act) v = siluf(v);
      C[(size_t)mrow * ldc + ncol] = v;
    }
  }
}

// ---------------- fused edge kernel (64 edges per block) ----------------
// Each wave owns an 80-column group (5 n-tiles); each B fragment load is reused
// across 4 M-tiles (4 WMMAs) to cut W2 L2 traffic 4x vs. 16-edge blocks.
#define EDGE_SMEM ((ETILE*128 + ETILE*32 + ETILE*640) * 2 + ETILE * 12)
__global__ __launch_bounds__(256) void k_edge(
    const float* __restrict__ st_in, float* __restrict__ st_out,
    const int* __restrict__ elist, const float* __restrict__ rbf,
    const float* __restrict__ cut, const float* __restrict__ evec,
    const __bf16* __restrict__ W1p, const float* __restrict__ b1,
    const __bf16* __restrict__ W2p, const float* __restrict__ b2,
    const __bf16* __restrict__ Wrp, const float* __restrict__ rb,
    int E) {
  extern __shared__ __align__(16) char smem[];
  __bf16* s_s = (__bf16*)smem;            // 64 x 128 sender scalar features
  __bf16* s_rbf = s_s + ETILE * 128;      // 64 x 32 rbf (padded 20->32)
  __bf16* s_h = s_rbf + ETILE * 32;       // 64 x 640 hidden after silu
  int* s_snd = (int*)(s_h + ETILE * 640);
  int* s_rcv = s_snd + ETILE;
  float* s_cut = (float*)(s_rcv + ETILE);

  int e0 = blockIdx.x * ETILE;
  int tid = threadIdx.x;
  if (tid < ETILE) {
    int e = e0 + tid;
    s_snd[tid] = (e < E) ? elist[2 * (size_t)e] : 0;
    s_rcv[tid] = (e < E) ? elist[2 * (size_t)e + 1] : 0;
    s_cut[tid] = (e < E) ? cut[e] : 0.f;
  }
  __syncthreads();
  // stage sender features: float4 loads, packed bf16 converts
  for (int i = tid * 4; i < ETILE * 128; i += 1024) {
    int r = i >> 7, c = i & 127;
    float4 v = *reinterpret_cast<const float4*>(st_in + (size_t)s_snd[r] * SD + c);
    __bf16* d = &s_s[i];
    d[0] = (__bf16)v.x; d[1] = (__bf16)v.y; d[2] = (__bf16)v.z; d[3] = (__bf16)v.w;
  }
  for (int i = tid; i < ETILE * 32; i += 256) {
    int r = i >> 5, c = i & 31;
    float v = (e0 + r < E && c < NRBF) ? rbf[(size_t)(e0 + r) * NRBF + c] : 0.f;
    s_rbf[i] = (__bf16)v;
  }
  __syncthreads();

  int wv = tid >> 5, lane = tid & 31;
  int kbase = (lane & 16) ? 8 : 0;
  int mrow_lo = (lane & 16) ? 8 : 0;
  int lrow = lane & 15;

  // ---- stage 1: h = silu(s @ W1 + b1); K=128 (4 ktiles), 5 ntiles x 4 mtiles per wave ----
#pragma unroll
  for (int nn = 0; nn < 5; ++nn) {
    int nt = wv * 5 + nn;
    v8f acc[4];
#pragma unroll
    for (int m = 0; m < 4; ++m) acc[m] = (v8f){};
#pragma unroll
    for (int kt = 0; kt < 4; ++kt) {
      v16bf b = *reinterpret_cast<const v16bf*>(W1p + ((size_t)nt * 4 + kt) * 512 + lane * 16);
#pragma unroll
      for (int m = 0; m < 4; ++m) {
        v16bf a = ldsfrag(&s_s[(m * 16 + lrow) * 128 + kt * 32 + kbase]);
        acc[m] = wmma_bf16(a, b, acc[m]);
      }
    }
    int ncol = nt * 16 + lrow;
    float bv = b1[ncol];
#pragma unroll
    for (int m = 0; m < 4; ++m)
#pragma unroll
      for (int i = 0; i < 8; ++i)
        s_h[(m * 16 + mrow_lo + i) * 640 + ncol] = (__bf16)siluf(acc[m][i] + bv);
  }
  __syncthreads();

  // ---- stage 2: rbf GEMM + phi GEMM + gating + scatter ----
#pragma unroll
  for (int nn = 0; nn < 5; ++nn) {
    int nt = wv * 5 + nn;
    // rbf_out tiles (K=32, one ktile)
    v8f r8m[4];
    {
      v16bf brb = *reinterpret_cast<const v16bf*>(Wrp + (size_t)nt * 512 + lane * 16);
#pragma unroll
      for (int m = 0; m < 4; ++m) {
        v16bf arb = ldsfrag(&s_rbf[(m * 16 + lrow) * 32 + kbase]);
        v8f z = (v8f){};
        r8m[m] = wmma_bf16(arb, brb, z);
      }
    }
    v8f acc[4];
#pragma unroll
    for (int m = 0; m < 4; ++m) acc[m] = (v8f){};
    for (int kt = 0; kt < 20; ++kt) {
      v16bf b = *reinterpret_cast<const v16bf*>(W2p + ((size_t)nt * 20 + kt) * 512 + lane * 16);
#pragma unroll
      for (int m = 0; m < 4; ++m) {
        v16bf a = ldsfrag(&s_h[(m * 16 + lrow) * 640 + kt * 32 + kbase]);
        acc[m] = wmma_bf16(a, b, acc[m]);
      }
    }
    int ncol = nt * 16 + lrow;
    int chunk = ncol >> 7;  // uniform per nn: 0:g_s 1:g_v 2:g_d 3:g_b 4:g_t
    int cc = ncol & 127;
    float b2v = b2[ncol], rbv = rb[ncol];
#pragma unroll
    for (int m = 0; m < 4; ++m) {
#pragma unroll
      for (int i = 0; i < 8; ++i) {
        int mrow = m * 16 + mrow_lo + i;
        int e = e0 + mrow;
        if (e >= E) continue;
        float g = (acc[m][i] + b2v) * ((r8m[m][i] + rbv) * s_cut[mrow]);
        int snd = s_snd[mrow], rcv = s_rcv[mrow];
        const float* si = st_in + (size_t)snd * SD;
        float* so = st_out + (size_t)rcv * SD;
        if (chunk == 0) {
          atomicAdd(so + cc, g);
        } else if (chunk == 1) {
#pragma unroll
          for (int ch = 0; ch < 3; ++ch)
            atomicAdd(so + (1 + ch) * DD + cc, g * si[(1 + ch) * DD + cc]);
        } else if (chunk == 2) {
#pragma unroll
          for (int ch = 0; ch < 3; ++ch)
            atomicAdd(so + (1 + ch) * DD + cc, g * evec[(size_t)e * 3 + ch]);
        } else if (chunk == 3) {
#pragma unroll
          for (int ch = 0; ch < 3; ++ch)
            atomicAdd(so + (4 + ch) * DD + cc, g * si[(4 + ch) * DD + cc]);
        } else {
          atomicAdd(so + 7 * DD + cc, g * si[7 * DD + cc]);
        }
      }
    }
  }
}

// ---------------- small elementwise kernels ----------------
__global__ void k_init_state(float* __restrict__ st, const int* __restrict__ an,
                             const float* __restrict__ es, const float* __restrict__ et, int N) {
  int t = blockIdx.x * 256 + threadIdx.x;
  int n = t >> 7, d = t & 127;
  if (n >= N) return;
  int a = an[n];
  float* s = st + (size_t)n * SD;
  s[d] = es[a * DD + d];
#pragma unroll
  for (int ch = 1; ch <= 6; ++ch) s[ch * DD + d] = 0.f;
  s[7 * DD + d] = et[a * DD + d];
}

__global__ void k_edge_prep(const float* __restrict__ len, float* __restrict__ rbf,
                            float* __restrict__ cut, int E) {
  int e = blockIdx.x * 256 + threadIdx.x;
  if (e >= E) return;
  float r = fmaxf(len[e], 1e-6f);
  const float PI = 3.14159265358979f;
  float inv = 1.f / r;
#pragma unroll
  for (int j = 0; j < NRBF; ++j)
    rbf[(size_t)e * NRBF + j] = __sinf(r * (float)(j + 1) * (PI / RCUTF)) * inv;
  float c = 0.5f * (__cosf(PI * r / RCUTF) + 1.f);
  cut[e] = (r < RCUTF) ? c : 0.f;
}

// weighted geometric product: Z = wgp(U, V, WZ)
__global__ void k_wgp(const float* __restrict__ U, const float* __restrict__ V,
                      const float* __restrict__ WZ, float* __restrict__ Z, int N) {
  static constexpr int GP[64] = {
      0, 1, 2, 3, 4, 5, 6, 7,  1, 0, 4, 14, 2, 7, 11, 5,
      2, 12, 0, 5, 9, 3, 7, 6, 3, 6, 13, 0, 7, 10, 1, 4,
      4, 10, 1, 7, 8, 14, 5, 11, 5, 7, 11, 2, 6, 8, 12, 9,
      6, 3, 7, 9, 13, 4, 8, 10, 7, 5, 6, 4, 11, 9, 10, 8};
  static constexpr int WI[64] = {
      0, 1, 1, 1, 2, 2, 2, 3,  4, 5, 6, 6, 7, 8, 7, 9,
      4, 6, 5, 6, 7, 7, 8, 9,  4, 6, 6, 5, 8, 7, 7, 9,
      10, 11, 11, 12, 13, 14, 14, 15, 10, 12, 11, 11, 14, 13, 14, 15,
      10, 11, 12, 11, 14, 14, 13, 15, 16, 17, 17, 17, 18, 18, 18, 19};
  int t = blockIdx.x * 256 + threadIdx.x;
  int n = t >> 7, d = t & 127;
  if (n >= N) return;
  const float* u = U + (size_t)n * SD + d;
  const float* v = V + (size_t)n * SD + d;
  float uu[8], vv[8], wv[20];
#pragma unroll
  for (int i = 0; i < 8; ++i) { uu[i] = u[i * DD]; vv[i] = v[i * DD]; }
#pragma unroll
  for (int j = 0; j < 20; ++j) wv[j] = WZ[j * DD + d];
  float comb[16];
#pragma unroll
  for (int i = 0; i < 16; ++i) comb[i] = 0.f;
#pragma unroll
  for (int i = 0; i < 8; ++i)
#pragma unroll
    for (int j = 0; j < 8; ++j)
      comb[GP[i * 8 + j]] += uu[i] * vv[j] * wv[WI[i * 8 + j]];
  float* z = Z + (size_t)n * SD + d;
#pragma unroll
  for (int i = 0; i < 8; ++i) z[i * DD] = comb[i] - comb[i + 8];
}

__global__ void k_vnorm_updin(const float* __restrict__ st, const float* __restrict__ V,
                              float* __restrict__ updin, int N) {
  int t = blockIdx.x * 256 + threadIdx.x;
  int n = t >> 7, d = t & 127;
  if (n >= N) return;
  float v1 = V[(size_t)n * SD + DD + d];
  float v2 = V[(size_t)n * SD + 2 * DD + d];
  float v3 = V[(size_t)n * SD + 3 * DD + d];
  updin[(size_t)n * 256 + d] = st[(size_t)n * SD + d];
  updin[(size_t)n * 256 + 128 + d] = sqrtf(v1 * v1 + v2 * v2 + v3 * v3);
}

__global__ void k_add3(float* __restrict__ a, const float* __restrict__ b,
                       const float* __restrict__ c, int n) {
  int i = blockIdx.x * 256 + threadIdx.x;
  if (i < n) a[i] += b[i] + c[i];
}

__global__ void k_maskadd(float* __restrict__ a, const float* __restrict__ t,
                          const int* __restrict__ an, int ty, int N) {
  int i = blockIdx.x * 256 + threadIdx.x;
  if (i >= N * 512) return;
  if (an[i >> 9] == ty) a[i] += t[i];
}

__global__ void k_update(float* __restrict__ st, const float* __restrict__ a,
                         const float* __restrict__ S, int N) {
  int t = blockIdx.x * 256 + threadIdx.x;
  int n = t >> 7, d = t & 127;
  if (n >= N) return;
  const float* an_ = a + (size_t)n * 512;
  float as = an_[d], av = an_[128 + d], ab = an_[256 + d], at = an_[384 + d];
  float* s = st + (size_t)n * SD;
  const float* Sv = S + (size_t)n * SD;
  s[d] += as * Sv[d];
#pragma unroll
  for (int ch = 0; ch < 3; ++ch) s[(1 + ch) * DD + d] += av * Sv[(1 + ch) * DD + d];
#pragma unroll
  for (int ch = 0; ch < 3; ++ch) s[(4 + ch) * DD + d] += ab * Sv[(4 + ch) * DD + d];
  s[7 * DD + d] += at * Sv[7 * DD + d];
}

__global__ void k_out(const float* __restrict__ st, const float* __restrict__ coord,
                      const int* __restrict__ gidx, float* __restrict__ out, int N) {
  int n = blockIdx.x * 256 + threadIdx.x;
  if (n >= N) return;
  const float* s0 = st + (size_t)n * SD;
  float s = 0.f;
  for (int d = 0; d < DD; d += 4) {
    float4 v = *reinterpret_cast<const float4*>(s0 + d);
    s += v.x + v.y + v.z + v.w;
  }
  float x = coord[(size_t)n * 3], y = coord[(size_t)n * 3 + 1], z = coord[(size_t)n * 3 + 2];
  atomicAdd(out + gidx[n], s * (x * x + y * y + z * z));
}

// ---------------- host ----------------
extern "C" void kernel_launch(void* const* d_in, const int* in_sizes, int n_in,
                              void* d_out, int out_size, void* d_ws, size_t ws_size,
                              hipStream_t stream) {
  (void)n_in; (void)ws_size;
  const int* an = (const int*)d_in[0];
  const int* el = (const int*)d_in[1];
  const float* elen = (const float*)d_in[2];
  const float* evec = (const float*)d_in[3];
  const float* coord = (const float*)d_in[4];
  const int* gidx = (const int*)d_in[5];
  const float* emb_s = (const float*)d_in[8];
  const float* emb_t = (const float*)d_in[9];
  const int N = in_sizes[0];
  const int E = in_sizes[2];
  auto RP = [&](int r, int j) -> const float* { return (const float*)d_in[10 + r * 32 + j]; };

  // bump allocator over workspace
  char* p = (char*)d_ws;
  auto alloc = [&](size_t bytes) -> void* {
    void* q = (void*)p;
    p += (bytes + 255) & ~(size_t)255;
    return q;
  };
  size_t NS = (size_t)N * SD * sizeof(float);
  float* stA = (float*)alloc(NS);
  float* stB = (float*)alloc(NS);
  float* bufU = (float*)alloc(NS);
  float* bufV = (float*)alloc(NS);
  float* bufZ = (float*)alloc(NS);
  float* bufL = (float*)alloc(NS);
  float* updin = (float*)alloc((size_t)N * 256 * sizeof(float));
  float* hbuf = (float*)alloc((size_t)N * 512 * sizeof(float));
  float* tbuf = (float*)alloc((size_t)N * 512 * sizeof(float));
  float* abuf = (float*)alloc((size_t)N * 512 * sizeof(float));
  float* rbfb = (float*)alloc((size_t)E * NRBF * sizeof(float));
  float* cutb = (float*)alloc((size_t)E * sizeof(float));
  const size_t RSTRIDE = 2740224;  // packed bf16 elements per round
  __bf16* wpack = (__bf16*)alloc(RSTRIDE * NROUNDS * sizeof(__bf16));

  // packed offsets (elements) within a round
  const size_t OFF_RBF = 0;        // 20x640   -> 1*40*512
  const size_t OFF_W1 = 20480;     // 128x640  -> 4*40*512
  const size_t OFF_W2 = 102400;    // 640x640  -> 20*40*512
  const size_t OFF_PROJ = 512000;  // 16x (128x128 -> 4*8*512=16384)
  const size_t OFF_U1 = 774144;    // 5x (256x512 -> 131072)
  const size_t OFF_U2 = 1429504;   // 5x (512x512 -> 262144)

  auto repack = [&](const float* src, __bf16* dst, int K, int Nc) {
    int Kt = (K + 31) / 32, Nt = (Nc + 15) / 16;
    int tot = Kt * Nt * 512;
    k_repack<<<(tot + 255) / 256, 256, 0, stream>>>(src, dst, K, Nc);
  };
  for (int r = 0; r < NROUNDS; ++r) {
    __bf16* wr = wpack + RSTRIDE * r;
    repack(RP(r, 0), wr + OFF_RBF, NRBF, H5);
    repack(RP(r, 2), wr + OFF_W1, DD, H5);
    repack(RP(r, 4), wr + OFF_W2, H5, H5);
    for (int pre = 0; pre < 4; ++pre) {
      int b0 = 6 + pre * 5;  // s_W, s_b, v_W, b_W, t_W
      repack(RP(r, b0 + 0), wr + OFF_PROJ + (size_t)(pre * 4 + 0) * 16384, DD, DD);
      repack(RP(r, b0 + 2), wr + OFF_PROJ + (size_t)(pre * 4 + 1) * 16384, DD, DD);
      repack(RP(r, b0 + 3), wr + OFF_PROJ + (size_t)(pre * 4 + 2) * 16384, DD, DD);
      repack(RP(r, b0 + 4), wr + OFF_PROJ + (size_t)(pre * 4 + 3) * 16384, DD, DD);
    }
    for (int t = 0; t < NTYPES; ++t) {
      repack(RP(r, 28) + (size_t)t * 256 * 512, wr + OFF_U1 + (size_t)t * 131072, 256, 512);
      repack(RP(r, 30) + (size_t)t * 512 * 512, wr + OFF_U2 + (size_t)t * 262144, 512, 512);
    }
  }

  k_init_state<<<(N * DD + 255) / 256, 256, 0, stream>>>(stA, an, emb_s, emb_t, N);
  k_edge_prep<<<(E + 255) / 256, 256, 0, stream>>>(elen, rbfb, cutb, E);

  auto gemm = [&](const float* A, int lda, const __bf16* W, const float* bias,
                  float* C, int ldc, int M, int K, int Nc, int act) {
    int Mt = (M + 15) / 16, Nt4 = Nc / 64;
    int tiles = Mt * Nt4;
    k_gemm<<<(tiles + 7) / 8, 256, 0, stream>>>(A, lda, W, bias, C, ldc, M, K, Nc, act);
  };

  float* cur = stA;
  float* nxt = stB;
  for (int r = 0; r < NROUNDS; ++r) {
    __bf16* wr = wpack + RSTRIDE * r;
    hipMemcpyAsync(nxt, cur, NS, hipMemcpyDeviceToDevice, stream);
    k_edge<<<(E + ETILE - 1) / ETILE, 256, EDGE_SMEM, stream>>>(
        cur, nxt, el, rbfb, cutb, evec,
        wr + OFF_W1, RP(r, 3), wr + OFF_W2, RP(r, 5),
        wr + OFF_RBF, RP(r, 1), E);
    auto proj = [&](const float* in, float* out, int pre, const float* s_bias) {
      const __bf16* base = wr + OFF_PROJ + (size_t)pre * 4 * 16384;
      gemm(in, SD, base, s_bias, out, SD, N, DD, DD, 0);
      for (int c = 1; c <= 3; ++c) gemm(in + c * DD, SD, base + 16384, nullptr, out + c * DD, SD, N, DD, DD, 0);
      for (int c = 4; c <= 6; ++c) gemm(in + c * DD, SD, base + 2 * 16384, nullptr, out + c * DD, SD, N, DD, DD, 0);
      gemm(in + 7 * DD, SD, base + 3 * 16384, nullptr, out + 7 * DD, SD, N, DD, DD, 0);
    };
    proj(nxt, bufU, 0, RP(r, 7));    // U
    proj(nxt, bufV, 1, RP(r, 12));   // V
    k_vnorm_updin<<<(N * DD + 255) / 256, 256, 0, stream>>>(nxt, bufV, updin, N);
    k_wgp<<<(N * DD + 255) / 256, 256, 0, stream>>>(bufU, bufV, RP(r, 26), bufZ, N);  // Z1
    proj(bufZ, bufL, 2, RP(r, 17));  // Z1_l
    k_wgp<<<(N * DD + 255) / 256, 256, 0, stream>>>(bufU, bufL, RP(r, 27), bufZ, N);  // Z2
    proj(bufZ, bufV, 3, RP(r, 22));  // Z2_l (V dead, reuse)
    k_add3<<<((int)((size_t)N * SD) + 255) / 256, 256, 0, stream>>>(bufU, bufL, bufV, (int)((size_t)N * SD));  // S in U
    hipMemsetAsync(abuf, 0, (size_t)N * 512 * sizeof(float), stream);
    for (int t = 0; t < NTYPES; ++t) {
      gemm(updin, 256, wr + OFF_U1 + (size_t)t * 131072, RP(r, 29) + t * 512, hbuf, 512, N, 256, 512, 1);
      gemm(hbuf, 512, wr + OFF_U2 + (size_t)t * 262144, RP(r, 31) + t * 512, tbuf, 512, N, 512, 512, 0);
      k_maskadd<<<(N * 512 + 255) / 256, 256, 0, stream>>>(abuf, tbuf, an, t, N);
    }
    k_update<<<(N * DD + 255) / 256, 256, 0, stream>>>(nxt, abuf, bufU, N);
    float* tmp = cur; cur = nxt; nxt = tmp;
  }

  hipMemsetAsync(d_out, 0, (size_t)out_size * sizeof(float), stream);
  k_out<<<(N + 255) / 256, 256, 0, stream>>>(cur, coord, gidx, (float*)d_out, N);
}
